// ScaledDotProductAttention_48172353192603
// MI455X (gfx1250) — matrix operand order repaired and checked by reading, not compiled, not run on hardware
//
#include <hip/hip_runtime.h>

// MI455X (gfx1250) fused attention:
//   q = (x @ Wq) * H^-0.5 ; k = x @ Wk ; v = x @ Wv   (f16 intermediates in d_ws)
//   out = softmax(mask(q k^T)) v, mask allows s >= t (strictly-lower tri = -inf)
// Flash-attention style, WMMA f16, TDM (tensor_load_to_lds) double-buffered V tiles.

#define TT 2048
#define EE 768
#define HH 64

typedef __attribute__((ext_vector_type(16))) _Float16 v16h;
typedef __attribute__((ext_vector_type(8)))  _Float16 h8;
typedef __attribute__((ext_vector_type(8)))  float    v8f;
typedef __attribute__((ext_vector_type(4)))  unsigned int u32x4;
typedef __attribute__((ext_vector_type(8)))  int i32x8;
typedef __attribute__((ext_vector_type(4)))  int i32x4;

__device__ __forceinline__ v8f wmma_f16(v16h a, v16h b, v8f c) {
  // D = A(16x32 f16) * B(32x16 f16) + C(16x16 f32)
  return __builtin_amdgcn_wmma_f32_16x16x32_f16(false, a, false, b, (short)0, c,
                                                false, false);
}

__device__ __forceinline__ v8f vzero() {
  v8f z;
#pragma unroll
  for (int i = 0; i < 8; ++i) z[i] = 0.f;
  return z;
}

// A-fragment (16x32 f16). Lane L holds row M = L&15. Element e -> K = 2*kb+e,
// kb = (L<16 ? 0 : 8). Two contiguous 16B loads.
__device__ __forceinline__ v16h fragA(const _Float16* base, int kbA) {
  h8 x0 = *(const h8*)(base + 2 * kbA);
  h8 x1 = *(const h8*)(base + 2 * kbA + 8);
  v16h r;
#pragma unroll
  for (int i = 0; i < 8; ++i) { r[i] = x0[i]; r[i + 8] = x1[i]; }
  return r;
}

// ---------------------------------------------------------------------------
// TDM: async DMA of a 2D f16 tile [rows=64 x cols=32] (row stride = TT elems)
// from global into LDS (contiguous [64][32]). tensor_dim0 = remaining columns
// so the hardware zero-fills out-of-range reads (tail chunk).
// D# packing per CDNA5 ISA ch.8 (group0 128b, group1 256b; groups 2/3 unused).
// This toolchain exposes the 6-arg builtin (extra i32x8 before cpol).
// ---------------------------------------------------------------------------
__device__ __forceinline__ void tdm_load_v(const _Float16* gbase,
                                           unsigned lds_byte_addr,
                                           unsigned remaining_cols) {
  unsigned long long ga = (unsigned long long)(const void*)gbase;
  u32x4 g0;
  g0[0] = 1u;                                     // count=1, user descriptor
  g0[1] = lds_byte_addr;                          // lds_addr [63:32]
  g0[2] = (unsigned)(ga & 0xffffffffu);           // global_addr[31:0]
  g0[3] = (unsigned)((ga >> 32) & 0x01ffffffu)    // global_addr[56:32]
        | (2u << 30);                             // type=2 ("image")

  const unsigned td0 = remaining_cols;            // tensor_dim0 (OOB -> zero)
  const unsigned td1 = HH;                        // tensor_dim1 = 64 rows
  const unsigned tile0 = 32, tile1 = HH;
  const unsigned long long st0 = (unsigned long long)TT;  // dim0 stride

  i32x8 g1;
  g1[0] = (int)(1u << 16);                        // wg_mask=0, data_size=1 (2B)
  g1[1] = (int)((td0 & 0xffffu) << 16);           // tensor_dim0[15:0] @ 63:48
  g1[2] = (int)(((td0 >> 16) & 0xffffu) | ((td1 & 0xffffu) << 16));
  g1[3] = (int)(((td1 >> 16) & 0xffffu) | (tile0 << 16));  // tile_dim0 @127:112
  g1[4] = (int)(tile1 & 0xffffu);                 // tile_dim1; tile_dim2=0
  g1[5] = (int)(unsigned)(st0 & 0xffffffffu);     // stride0[31:0]
  g1[6] = (int)(unsigned)((st0 >> 32) & 0xffffu); // stride0[47:32]; stride1=0
  g1[7] = 0;

  i32x4 z4;
  z4[0] = z4[1] = z4[2] = z4[3] = 0;
  i32x8 z8;
#pragma unroll
  for (int i = 0; i < 8; ++i) z8[i] = 0;
  __builtin_amdgcn_tensor_load_to_lds(g0, g1, z4, z4, z8, 0);
}

// ---------------------------------------------------------------------------
// Kernel 0: one-time weight prep: fp32 [E,H] -> f16 transposed [3][H][E]
// (K-contiguous so B-fragments are single 32B loads; 288KB, L2-resident).
// ---------------------------------------------------------------------------
__global__ __launch_bounds__(256) void prep_w(const float* __restrict__ Wq,
                                              const float* __restrict__ Wk,
                                              const float* __restrict__ Wv,
                                              _Float16* __restrict__ wt) {
  int idx = blockIdx.x * 256 + threadIdx.x;
  if (idx >= 3 * HH * EE) return;
  int w   = idx / (HH * EE);
  int rem = idx - w * (HH * EE);
  int h   = rem / EE;
  int e   = rem - h * EE;
  const float* W = (w == 0) ? Wq : (w == 1) ? Wk : Wv;
  wt[idx] = (_Float16)W[e * HH + h];
}

// ---------------------------------------------------------------------------
// Kernel 1: QKV projection. One wave per 16-row x tile; no LDS, no barriers:
// all 12 B-fragments fetched as clause-able global loads, then 12 WMMAs.
// ---------------------------------------------------------------------------
__global__ __launch_bounds__(128) void qkv_proj(
    const float* __restrict__ x, const _Float16* __restrict__ wt,
    _Float16* __restrict__ qh, _Float16* __restrict__ kh,
    _Float16* __restrict__ vt) {
  const int lane = threadIdx.x & 31;
  const int wv   = threadIdx.x >> 5;
  const int tile = blockIdx.x * 4 + wv;
  const int row0 = tile << 4;  // global row index (b*T + t)
  const bool lo  = lane < 16;
  const int  ln   = lane & 15;
  const int  kbA  = lo ? 0 : 8;
  const int  koff = lo ? 0 : 16;
  const int  mhi  = lo ? 0 : 8;

  const _Float16* wq = wt;
  const _Float16* wk = wt + (size_t)HH * EE;
  const _Float16* wvp = wt + (size_t)2 * HH * EE;

  v8f aq[4], ak[4], av[4];
#pragma unroll
  for (int nt = 0; nt < 4; ++nt) { aq[nt] = vzero(); ak[nt] = vzero(); av[nt] = vzero(); }

  for (int e0 = 0; e0 < EE; e0 += 32) {
    // A fragment of x (fp32 -> f16)
    const float* xr = x + (size_t)(row0 + ln) * EE + e0;
    float4 u0 = *(const float4*)(xr + koff);
    float4 u1 = *(const float4*)(xr + koff + 4);
    float4 u2 = *(const float4*)(xr + koff + 8);
    float4 u3 = *(const float4*)(xr + koff + 12);

    // All B fragments first (contiguous 32B loads, compiler can clause these)
    v16h bq[4], bk[4], bv[4];
#pragma unroll
    for (int nt = 0; nt < 4; ++nt) {
      size_t wo = (size_t)(nt * 16 + ln) * EE + e0 + koff;
      bq[nt] = *(const v16h*)(wq + wo);
      bk[nt] = *(const v16h*)(wk + wo);
      bv[nt] = *(const v16h*)(wvp + wo);
    }

    v16h a;
    a[0]=(_Float16)u0.x; a[1]=(_Float16)u0.y; a[2]=(_Float16)u0.z; a[3]=(_Float16)u0.w;
    a[4]=(_Float16)u1.x; a[5]=(_Float16)u1.y; a[6]=(_Float16)u1.z; a[7]=(_Float16)u1.w;
    a[8]=(_Float16)u2.x; a[9]=(_Float16)u2.y; a[10]=(_Float16)u2.z; a[11]=(_Float16)u2.w;
    a[12]=(_Float16)u3.x; a[13]=(_Float16)u3.y; a[14]=(_Float16)u3.z; a[15]=(_Float16)u3.w;

#pragma unroll
    for (int nt = 0; nt < 4; ++nt) {
      aq[nt] = wmma_f16(a, bq[nt], aq[nt]);
      ak[nt] = wmma_f16(a, bk[nt], ak[nt]);
      av[nt] = wmma_f16(a, bv[nt], av[nt]);
    }
  }

  // C layout: lanes 0-15 hold rows r, lanes 16-31 rows r+8; col = lane&15.
  const int b  = row0 >> 11;  // T = 2048
  const int t0 = row0 & (TT - 1);
#pragma unroll
  for (int nt = 0; nt < 4; ++nt)
#pragma unroll
    for (int r = 0; r < 8; ++r) {
      int h = nt * 16 + ln;
      int mm = r + mhi;
      size_t rg = (size_t)(row0 + mm);
      qh[rg * HH + h] = (_Float16)(aq[nt][r] * 0.125f);  // fold 1/sqrt(64)
      kh[rg * HH + h] = (_Float16)(ak[nt][r]);
      vt[((size_t)(b * HH + h)) * TT + (t0 + mm)] = (_Float16)(av[nt][r]);
    }
}

// ---------------------------------------------------------------------------
// Kernel 2: flash attention over allowed keys s >= t. One wave per 16 query
// rows; key chunks of 32 starting at the diagonal tile. V tiles arrive via
// TDM (tensor_load_to_lds), double-buffered one chunk ahead.
// ---------------------------------------------------------------------------
__global__ __launch_bounds__(64) void attn(
    const _Float16* __restrict__ qh, const _Float16* __restrict__ kh,
    const _Float16* __restrict__ vt, float* __restrict__ out) {
  __shared__ _Float16 sP[2][16][40];      // per-wave prob tile (C->A relayout)
  __shared__ _Float16 sV[2][2][HH][32];   // per-wave double-buffered V tiles

  const int lane = threadIdx.x & 31;
  const int wv   = threadIdx.x >> 5;
  const int tile = blockIdx.x * 2 + wv;
  const int b    = tile >> 7;             // T/16 = 128 tiles per batch
  const int t0   = (tile & 127) << 4;
  const int bT   = b * TT;
  const bool lo  = lane < 16;
  const int  ln   = lane & 15;
  const int  kbA  = lo ? 0 : 8;
  const int  koff = lo ? 0 : 16;
  const int  mhi  = lo ? 0 : 8;

  const _Float16* vbase = vt + (size_t)b * HH * TT;
  const unsigned ldsV0 = (unsigned)(unsigned long long)(uintptr_t)&sV[wv][0][0][0];
  const unsigned ldsV1 = (unsigned)(unsigned long long)(uintptr_t)&sV[wv][1][0][0];

  // q A-fragments over K = H (kept in registers for the whole pass)
  const _Float16* qrow = qh + (size_t)(bT + t0 + ln) * HH;
  const v16h qA0 = fragA(qrow, kbA);
  const v16h qA1 = fragA(qrow + 32, kbA);

  float Mr[8], Lr[8];
  v8f O[4];
#pragma unroll
  for (int r = 0; r < 8; ++r) { Mr[r] = -1e30f; Lr[r] = 0.f; }
#pragma unroll
  for (int nt = 0; nt < 4; ++nt) O[nt] = vzero();

  // Prime the V pipeline: chunk 0 -> buffer 0
  tdm_load_v(vbase + t0, ldsV0, (unsigned)(TT - t0));

  int cur = 0;
  for (int s0 = t0; s0 < TT; s0 += 32) {
    const bool more = (s0 + 32 < TT);
    if (more)  // prefetch next chunk's V tile into the other buffer
      tdm_load_v(vbase + (s0 + 32), cur ? ldsV0 : ldsV1,
                 (unsigned)(TT - (s0 + 32)));

    // ---- scores: S = q k^T for 32 keys (two 16x16 tiles) ----
    v8f S0 = vzero(), S1 = vzero();
    {
      int key0 = s0 + ln;  // always < T
      const _Float16* kr0 = kh + (size_t)(bT + key0) * HH;
      S0 = wmma_f16(qA0, *(const v16h*)(kr0 + koff), S0);
      S0 = wmma_f16(qA1, *(const v16h*)(kr0 + 32 + koff), S0);
      int key1  = s0 + 16 + ln;
      int key1c = key1 < TT ? key1 : TT - 1;
      const _Float16* kr1 = kh + (size_t)(bT + key1c) * HH;
      S1 = wmma_f16(qA0, *(const v16h*)(kr1 + koff), S1);
      S1 = wmma_f16(qA1, *(const v16h*)(kr1 + 32 + koff), S1);
      if (more)
        __builtin_prefetch(kh + (size_t)(bT + s0 + 32 + ln) * HH, 0, 1);
    }
    // diagonal tile mask: drop s < t (only first chunk's first tile)
    if (s0 == t0) {
#pragma unroll
      for (int r = 0; r < 8; ++r)
        if (s0 + ln < t0 + r + mhi) S0[r] = -1e30f;
    }
    // out-of-range keys in second tile (tail chunk)
    if (s0 + 16 + ln >= TT) {
#pragma unroll
      for (int r = 0; r < 8; ++r) S1[r] = -1e30f;
    }

    // ---- online softmax (row stats shared across each 16-lane half) ----
#pragma unroll
    for (int r = 0; r < 8; ++r) {
      float mx = fmaxf(S0[r], S1[r]);
#pragma unroll
      for (int off = 1; off < 16; off <<= 1)
        mx = fmaxf(mx, __shfl_xor(mx, off, 32));
      float Mn    = fmaxf(Mr[r], mx);
      float alpha = __expf(Mr[r] - Mn);
      float p0    = __expf(S0[r] - Mn);
      float p1    = __expf(S1[r] - Mn);
      float rs    = p0 + p1;
#pragma unroll
      for (int off = 1; off < 16; off <<= 1)
        rs += __shfl_xor(rs, off, 32);
      Lr[r] = Lr[r] * alpha + rs;
      Mr[r] = Mn;
#pragma unroll
      for (int nt = 0; nt < 4; ++nt) O[nt][r] *= alpha;
      int mm = r + mhi;
      sP[wv][mm][ln]      = (_Float16)p0;
      sP[wv][mm][16 + ln] = (_Float16)p1;
    }
    __builtin_amdgcn_wave_barrier();  // LDS ops are in-order per wave

    // ---- P (A-layout via LDS) x V (TDM-staged LDS tile) ----
    v16h pA = fragA(&sP[wv][ln][0], kbA);
    if (more) __builtin_amdgcn_s_wait_tensorcnt(1);  // current tile landed
    else      __builtin_amdgcn_s_wait_tensorcnt(0);
#pragma unroll
    for (int nt = 0; nt < 4; ++nt) {
      int h = nt * 16 + ln;
      v16h vB = *(const v16h*)&sV[wv][cur][h][koff];  // OOB cols are HW zeros
      O[nt] = wmma_f16(pA, vB, O[nt]);
    }
    cur ^= 1;
  }

  // ---- normalize and store fp32 output ----
#pragma unroll
  for (int r = 0; r < 8; ++r) {
    float inv = 1.0f / Lr[r];  // L > 0: diagonal key is always allowed
    int mm = r + mhi;
    float* orow = out + (size_t)(bT + t0 + mm) * HH + ln;
    orow[0]  = O[0][r] * inv;
    orow[16] = O[1][r] * inv;
    orow[32] = O[2][r] * inv;
    orow[48] = O[3][r] * inv;
  }
}

// ---------------------------------------------------------------------------
extern "C" void kernel_launch(void* const* d_in, const int* in_sizes, int n_in,
                              void* d_out, int out_size, void* d_ws,
                              size_t ws_size, hipStream_t stream) {
  const float* x  = (const float*)d_in[0];
  const float* Wq = (const float*)d_in[1];
  const float* Wk = (const float*)d_in[2];
  const float* Wv = (const float*)d_in[3];
  float* out = (float*)d_out;

  const int B = in_sizes[0] / (TT * EE);            // 4
  const size_t BTH = (size_t)B * TT * HH;
  _Float16* qh = (_Float16*)d_ws;                   // [B,T,H] f16 (pre-scaled)
  _Float16* kh = qh + BTH;                          // [B,T,H] f16
  _Float16* vt = kh + BTH;                          // [B,H,T] f16 (transposed)
  _Float16* wt = vt + BTH;                          // [3][H][E] f16 weights
  // workspace: 3*B*T*H*2 + 3*H*E*2 bytes = 3 MiB + 288 KiB

  prep_w<<<(3 * HH * EE + 255) / 256, 256, 0, stream>>>(Wq, Wk, Wv, wt);

  const int tiles = B * (TT / 16);                  // 512 query tiles
  qkv_proj<<<tiles / 4, 128, 0, stream>>>(x, wt, qh, kh, vt);
  attn<<<tiles / 2, 64, 0, stream>>>(qh, kh, vt, out);

  (void)n_in; (void)out_size; (void)ws_size;
}
